// QuantLinearW8A8_56478819942550
// MI455X (gfx1250) — matrix operand order repaired and checked by reading, not compiled, not run on hardware
//
#include <hip/hip_runtime.h>
#include <stdint.h>

typedef __attribute__((ext_vector_type(16))) _Float16 v16h;
typedef __attribute__((ext_vector_type(8)))  _Float16 v8h;
typedef __attribute__((ext_vector_type(2)))  _Float16 h2;
typedef __attribute__((ext_vector_type(8)))  float    v8f;
typedef __attribute__((ext_vector_type(4)))  float    v4f;
typedef __attribute__((ext_vector_type(4)))  uint32_t v4u;

constexpr int BM  = 128;
constexpr int BN  = 128;
constexpr int BK  = 64;            // two V_WMMA_F32_16X16X32_F16 K-steps per LDS stage
constexpr int LDA = BK + 8;        // 72 halves = 144 B row stride (16B-aligned, bank-staggered)
constexpr int LDB = BK + 8;

union Frag16 { v16h v; v8h h[2]; };
union H8     { v8h  v; uint32_t u[4]; };

__global__ __launch_bounds__(256)
void qlinear_w8_wmma_f16(const float* __restrict__ X,
                         const int8_t* __restrict__ Wq,
                         const float* __restrict__ scale,
                         const float* __restrict__ bias,
                         float* __restrict__ Y,
                         int M, int N, int K) {
  __shared__ _Float16 As[2][BM * LDA];   // activations, dequant-scaled, f16
  __shared__ _Float16 Bs[2][BN * LDB];   // weights, int8 -> f16 (exact)

  const int tid   = threadIdx.x;
  const int lane  = tid & 31;
  const int wave  = tid >> 5;            // 8 waves
  const int waveM = wave & 3;            // 4 waves along M (32 rows each)
  const int waveN = wave >> 2;           // 2 waves along N (64 cols each)

  const int blockM = blockIdx.y * BM;
  const int blockN = blockIdx.x * BN;

  // -------- global -> LDS staging assignment --------
  // A: thread loads 8 consecutive f32 for rows rowA + r*32, r = 0..3
  const int rowA = tid >> 3;             // 0..31
  const int kA0  = (tid & 7) * 8;        // 0..56
  // B: thread loads 16 int8 for rows rowB + r*64, r = 0..1
  const int rowB = tid >> 2;             // 0..63
  const int kB0  = (tid & 3) * 16;       // 0..48

  const float*  Xa = X  + (size_t)(blockM + rowA) * K + kA0;
  const int8_t* Wb = Wq + (size_t)(blockN + rowB) * K + kB0;

  const int nkt = K / BK;                // 64 stages

  // staging registers (global tile held here while previous tile computes)
  v8h ha[4];
  v8h bl[2], bh[2];

  // int8 -> f16 via bit trick:  f16(0x6400 | (q^0x80)) == 1024 + (q+128)  (exact),
  // then subtract 1152.  5 VALU ops per 4 elements (xor + 2 perm + 2 pk_add).
  const h2 kBias = { (_Float16)(-1152.0f), (_Float16)(-1152.0f) };
  auto cvt_dword = [&](uint32_t w, uint32_t& lo, uint32_t& hi) {
    uint32_t t  = w ^ 0x80808080u;
    uint32_t p0 = __builtin_amdgcn_perm(0x64646464u, t, 0x04010400u); // {0x64,t1,0x64,t0}
    uint32_t p1 = __builtin_amdgcn_perm(0x64646464u, t, 0x04030402u); // {0x64,t3,0x64,t2}
    h2 q0 = __builtin_bit_cast(h2, p0) + kBias;
    h2 q1 = __builtin_bit_cast(h2, p1) + kBias;
    lo = __builtin_bit_cast(uint32_t, q0);
    hi = __builtin_bit_cast(uint32_t, q1);
  };

  auto load_tile = [&](int kt) {
    const int gk = kt * BK;
    // per-in-channel scale folded into activations; reused across 4 row passes
    v4f s0 = *(const v4f*)(scale + gk + kA0);
    v4f s1 = *(const v4f*)(scale + gk + kA0 + 4);
#pragma unroll
    for (int r = 0; r < 4; ++r) {
      const float* p = Xa + gk + (size_t)(r * 32) * K;
      v4f a0 = *(const v4f*)p;
      v4f a1 = *(const v4f*)(p + 4);
      a0 *= s0; a1 *= s1;
#pragma unroll
      for (int i = 0; i < 4; ++i) {
        ha[r][i]     = (_Float16)a0[i];
        ha[r][4 + i] = (_Float16)a1[i];
      }
    }
    // int8 weights -> f16 (exact for |q| <= 127)
#pragma unroll
    for (int r = 0; r < 2; ++r) {
      v4u w = *(const v4u*)(Wb + gk + (size_t)(r * 64) * K);
      H8 lo, hi;
      cvt_dword(w.x, lo.u[0], lo.u[1]);
      cvt_dword(w.y, lo.u[2], lo.u[3]);
      cvt_dword(w.z, hi.u[0], hi.u[1]);
      cvt_dword(w.w, hi.u[2], hi.u[3]);
      bl[r] = lo.v;
      bh[r] = hi.v;
    }
  };

  auto store_tile = [&](int buf) {
#pragma unroll
    for (int r = 0; r < 4; ++r)
      *(v8h*)&As[buf][(rowA + r * 32) * LDA + kA0] = ha[r];
#pragma unroll
    for (int r = 0; r < 2; ++r) {
      *(v8h*)&Bs[buf][(rowB + r * 64) * LDB + kB0]     = bl[r];
      *(v8h*)&Bs[buf][(rowB + r * 64) * LDB + kB0 + 8] = bh[r];
    }
  };

  // -------- per-lane fragment coordinates (row/col = l15; contiguous k order k = 16*lhi + s for A and B) --------
  const int l15  = lane & 15;
  const int lhi  = lane >> 4;
  const int arow = waveM * 32 + l15;   // A row within block tile
  const int ak   = lhi * 16;           // lane<16: K 0-15 ; lane>=16: K 16-31
  const int brow = waveN * 64 + l15;   // B column (= output col) within block tile
  const int bk   = lhi * 16;           // lane<16: K 0-15 ; lane>=16: K 16-31

  v8f acc[2][4];
#pragma unroll
  for (int i = 0; i < 2; ++i)
#pragma unroll
    for (int j = 0; j < 4; ++j)
      acc[i][j] = {};

  // -------- prologue --------
  load_tile(0);
  store_tile(0);
  __syncthreads();

  int buf = 0;
  for (int kt = 0; kt < nkt; ++kt) {
    const bool has_next = (kt + 1) < nkt;
    if (has_next) {
      load_tile(kt + 1);                       // global loads overlap WMMAs below
      if (kt + 2 < nkt) {                      // warm L2 two stages ahead
        __builtin_prefetch(Xa + (size_t)(kt + 2) * BK, 0, 1);
        __builtin_prefetch(Wb + (size_t)(kt + 2) * BK, 0, 1);
      }
    }

    const _Float16* Ab = &As[buf][0];
    const _Float16* Bb = &Bs[buf][0];

    // two WMMA K-steps per staged tile -> 16 WMMAs per barrier per wave
#pragma unroll
    for (int ks = 0; ks < 2; ++ks) {
      const int kb = ks * 32;

      Frag16 a[2], b[4];
#pragma unroll
      for (int i = 0; i < 2; ++i) {
        const _Float16* p = Ab + (arow + i * 16) * LDA + kb + ak;
        a[i].h[0] = *(const v8h*)p;            // K ak..ak+7
        a[i].h[1] = *(const v8h*)(p + 8);      // K ak+8..ak+15
      }
#pragma unroll
      for (int j = 0; j < 4; ++j) {
        const _Float16* p = Bb + (brow + j * 16) * LDB + kb + bk;
        b[j].h[0] = *(const v8h*)p;            // K bk..bk+7
        b[j].h[1] = *(const v8h*)(p + 8);      // K bk+8..bk+15
      }

#pragma unroll
      for (int i = 0; i < 2; ++i)
#pragma unroll
        for (int j = 0; j < 4; ++j)
          acc[i][j] = __builtin_amdgcn_wmma_f32_16x16x32_f16(
              /*neg_a=*/false, a[i].v, /*neg_b=*/false, b[j].v,
              /*c_mod=*/(short)0, acc[i][j],
              /*reuse_a=*/false, /*reuse_b=*/false);
    }

    if (has_next) {
      store_tile(buf ^ 1);   // write the buffer nobody reads this iteration
      __syncthreads();       // one barrier per stage (64 total)
      buf ^= 1;
    }
  }

  // -------- epilogue: bias + store (C/D layout: VGPR r, lane -> M=lhi*8+r, N=l15) --------
#pragma unroll
  for (int j = 0; j < 4; ++j) {
    const int gcol = blockN + brow + j * 16;
    const float bv = bias[gcol];
#pragma unroll
    for (int i = 0; i < 2; ++i) {
      const int gm0 = blockM + waveM * 32 + i * 16 + lhi * 8;
#pragma unroll
      for (int r = 0; r < 8; ++r) {
        Y[(size_t)(gm0 + r) * N + gcol] = acc[i][j][r] + bv;
      }
    }
  }
}

extern "C" void kernel_launch(void* const* d_in, const int* in_sizes, int n_in,
                              void* d_out, int out_size, void* d_ws, size_t ws_size,
                              hipStream_t stream) {
  const float*  X  = (const float*)d_in[0];   // [B*S, IN_F] f32
  const int8_t* Wq = (const int8_t*)d_in[1];  // [OUT_F, IN_F] int8
  const float*  sc = (const float*)d_in[2];   // [IN_F] f32
  const float*  bs = (const float*)d_in[3];   // [OUT_F] f32
  float* Y = (float*)d_out;                   // [B*S, OUT_F] f32

  const int K = in_sizes[2];          // IN_F  = 4096
  const int N = in_sizes[3];          // OUT_F = 4096
  const int M = in_sizes[0] / K;      // B*S   = 8192

  dim3 grid(N / BN, M / BM);          // (32, 64)
  qlinear_w8_wmma_f16<<<grid, dim3(256), 0, stream>>>(X, Wq, sc, bs, Y, M, N, K);
}
